// CausalSelfAttention_78211354460161
// MI455X (gfx1250) — matrix operand order repaired and checked by reading, not compiled, not run on hardware
//
#include <hip/hip_runtime.h>
#include <hip/hip_bf16.h>

typedef __attribute__((ext_vector_type(16))) __bf16 v16bf;
typedef __attribute__((ext_vector_type(8)))  float  v8f;
typedef unsigned short ush;

__device__ __forceinline__ ush f2bf(float f) {
  unsigned u = __builtin_bit_cast(unsigned, f);
  u += 0x7FFFu + ((u >> 16) & 1u);          // round-to-nearest-even
  return (ush)(u >> 16);
}

union FragU { v16bf v; unsigned u[8]; };

// ---- CDNA5 async global->LDS DMA (ASYNCcnt), 16B per lane ----
__device__ __forceinline__ void async_ld_b128(unsigned lds_addr, const void* gp) {
  asm volatile("global_load_async_to_lds_b128 %0, %1, off"
               :: "v"(lds_addr), "v"(reinterpret_cast<unsigned long long>(gp))
               : "memory");
}
__device__ __forceinline__ void wait_async0() {
  asm volatile("s_wait_asynccnt 0x0" ::: "memory");
}
__device__ __forceinline__ unsigned lds_addr_of(const void* p) {
  return (unsigned)reinterpret_cast<unsigned long long>(p);   // addr[31:0] = LDS offset
}

// A-matrix 16x32 bf16 fragment (ISA 7.12.2): lanes 0-15 row M=lane hold K=0..7,16..23;
// lanes 16-31 row M=lane-16 hold K=8..15,24..31. Pairs packed per VGPR.
__device__ __forceinline__ v16bf load_frag_a(const ush* p0, int ldw, int row0, int k0) {
  const int lane = threadIdx.x & 31;
  const int half = lane >> 4;
  const ush* p = p0 + (row0 + (lane & 15)) * ldw + k0;
  FragU f;
#pragma unroll
  for (int i = 0; i < 8; ++i) {
    int k = ((i < 4) ? 0 : 16) + half * 8 + (i & 3) * 2;
    f.u[i] = *(const unsigned*)(p + k);
  }
  return f.v;
}

// B-matrix 32x16 bf16 fragment: lane = column N; lanes 0-15 hold K=0..7,16..23,
// lanes 16-31 hold K=8..15,24..31 (same order as A).
__device__ __forceinline__ v16bf load_frag_b(const ush* p0, int ldw, int row0, int k0) {
  const int lane = threadIdx.x & 31;
  const int half = lane >> 4;
  const ush* p = p0 + (row0 + (lane & 15)) * ldw + k0;
  FragU f;
#pragma unroll
  for (int i = 0; i < 8; ++i) f.u[i] = *(const unsigned*)(p + ((i < 4) ? 0 : 16) + half * 8 + (i & 3) * 2);
  return f.v;
}

// ---------------- conversion kernels ----------------

__global__ void cvt_f32_bf16(const float* __restrict__ in, ush* __restrict__ out, int n) {
  for (int i = blockIdx.x * blockDim.x + threadIdx.x; i < n; i += gridDim.x * blockDim.x)
    out[i] = f2bf(in[i]);
}

// W[k][n] (1024x1024) -> Wt[n][k] bf16
__global__ void transpose_cvt(const float* __restrict__ W, ush* __restrict__ Wt) {
  int idx = blockIdx.x * blockDim.x + threadIdx.x;   // 1M threads
  int n = idx >> 10, k = idx & 1023;
  Wt[idx] = f2bf(W[k * 1024 + n]);
}

// ---------------- bf16 WMMA GEMM: C[4096,1024] = A[4096,1024] * Bt[1024,1024]^T + bias ----

#define LDT 40   // K tile 32 + 8 pad (ush)

__device__ __forceinline__ void gemm_prefetch(const ush* __restrict__ A,
                                              const ush* __restrict__ Bt,
                                              ush* As, ush* Bs,
                                              int m_blk, int n_blk, int kk,
                                              int lr, int lh) {
  const ush* ga = A + (size_t)(m_blk + lr) * 1024 + kk + lh * 16;
  unsigned la = lds_addr_of(As + lr * LDT + lh * 16);
  async_ld_b128(la, ga);
  async_ld_b128(la + 16, ga + 8);
  const ush* gb = Bt + (size_t)(n_blk + lr) * 1024 + kk + lh * 16;
  unsigned lb = lds_addr_of(Bs + lr * LDT + lh * 16);
  async_ld_b128(lb, gb);
  async_ld_b128(lb + 16, gb + 8);
}

__device__ __forceinline__ void gemm_compute(const ush* As, const ush* Bs,
                                             int wm, int wn, v8f (&acc)[2][4]) {
  v16bf afr[2], bfr[4];
#pragma unroll
  for (int i = 0; i < 2; ++i) afr[i] = load_frag_a(As, LDT, wm * 32 + i * 16, 0);
#pragma unroll
  for (int j = 0; j < 4; ++j) bfr[j] = load_frag_b(Bs, LDT, wn * 64 + j * 16, 0);
#pragma unroll
  for (int i = 0; i < 2; ++i)
#pragma unroll
    for (int j = 0; j < 4; ++j)
      acc[i][j] = __builtin_amdgcn_wmma_f32_16x16x32_bf16(
          false, afr[i], false, bfr[j], (short)0, acc[i][j], false, false);
}

template <bool WF, bool WB>
__global__ __launch_bounds__(256) void gemm_bf16(
    const ush* __restrict__ A, const ush* __restrict__ Bt,
    const float* __restrict__ bias, float* __restrict__ Cf, ush* __restrict__ Cb) {
  alignas(16) __shared__ ush As[2][128 * LDT];
  alignas(16) __shared__ ush Bs[2][128 * LDT];
  const int t = threadIdx.x;
  const int lane = t & 31;
  const int wave = t >> 5;
  const int wm = wave >> 1, wn = wave & 1;        // 4x2 wave grid
  const int m_blk = blockIdx.y * 128;
  const int n_blk = blockIdx.x * 128;
  const int lr = t >> 1;    // 0..127 tile row
  const int lh = t & 1;     // which 16-wide half of the 32 K-slice

  v8f acc[2][4] = {};

  gemm_prefetch(A, Bt, As[0], Bs[0], m_blk, n_blk, 0, lr, lh);
  const int NK = 1024 / 32;            // 32 steps, even
  for (int ik = 0; ik < NK; ik += 2) {
    // even step: consume buffer 0, prefetch into buffer 1
    wait_async0();
    __syncthreads();
    if (ik + 1 < NK) gemm_prefetch(A, Bt, As[1], Bs[1], m_blk, n_blk, (ik + 1) * 32, lr, lh);
    gemm_compute(As[0], Bs[0], wm, wn, acc);
    // odd step: consume buffer 1, prefetch into buffer 0
    wait_async0();
    __syncthreads();
    if (ik + 2 < NK) gemm_prefetch(A, Bt, As[0], Bs[0], m_blk, n_blk, (ik + 2) * 32, lr, lh);
    gemm_compute(As[1], Bs[1], wm, wn, acc);
  }

  const int half = lane >> 4;
  const int nn = lane & 15;
#pragma unroll
  for (int i = 0; i < 2; ++i)
#pragma unroll
    for (int j = 0; j < 4; ++j) {
      const int col = n_blk + wn * 64 + j * 16 + nn;
      const float bv = bias[col];
      const int row0 = m_blk + wm * 32 + i * 16 + half * 8;
#pragma unroll
      for (int v = 0; v < 8; ++v) {
        float val = acc[i][j][v] + bv;
        size_t idx = (size_t)(row0 + v) * 1024 + col;
        if constexpr (WF) Cf[idx] = val;
        if constexpr (WB) Cb[idx] = f2bf(val);
      }
    }
}

// ---------------- flash attention (per b,h; 64 query rows / block; 4 waves) ----------------

#define LDQ 72    // 64 + 8 pad
#define LDV 136   // 128 + 8 pad

__global__ __launch_bounds__(128) void attn_kernel(
    const ush* __restrict__ Q, const ush* __restrict__ Kc,
    const ush* __restrict__ Vc, ush* __restrict__ Y) {
  alignas(16) __shared__ ush Qs[64 * LDQ];
  alignas(16) __shared__ ush Ks[128 * LDQ];
  alignas(16) __shared__ ush Vs[64 * LDV];          // transposed: [d][key]
  alignas(16) __shared__ ush Ps[4 * 16 * LDV];      // per-wave P (16 x 128)

  const int t = threadIdx.x;
  const int wave = t >> 5;
  const int lane = t & 31;
  const int half = lane >> 4;
  const int nn = lane & 15;
  const int b = blockIdx.z, h = blockIdx.y;
  const int q0 = blockIdx.x * 64;
  const size_t base = ((size_t)b * 2048) * 1024 + h * 64;

  {   // load Q tile 64x64
    int row = t >> 1, lh2 = t & 1;
    const uint4* g = (const uint4*)(Q + base + (size_t)(q0 + row) * 1024 + lh2 * 32);
    uint4* l = (uint4*)(Qs + row * LDQ + lh2 * 32);
    l[0] = g[0]; l[1] = g[1]; l[2] = g[2]; l[3] = g[3];
  }

  float mrun[8], lsum[8];
#pragma unroll
  for (int v = 0; v < 8; ++v) { mrun[v] = -1e30f; lsum[v] = 0.0f; }
  v8f o[4] = {};

  const int kbmax = (q0 + 63) >> 7;
  for (int kb = 0; kb <= kbmax; ++kb) {
    __syncthreads();   // all reads of Ks/Vs from previous iteration are done
    {   // K tile 128x64 row-major: async DMA straight into LDS (8x b128 per thread)
      const ush* g = Kc + base + (size_t)(kb * 128 + t) * 1024;
      unsigned l = lds_addr_of(Ks + t * LDQ);
#pragma unroll
      for (int i = 0; i < 8; ++i) async_ld_b128(l + i * 16, g + i * 8);
    }
    {   // V tile -> transposed Vs[d][key] (VGPR round-trip needed for transpose)
      const uint4* g = (const uint4*)(Vc + base + (size_t)(kb * 128 + t) * 1024);
      union { uint4 q4[8]; ush s[64]; } tmp;
#pragma unroll
      for (int i = 0; i < 8; ++i) tmp.q4[i] = g[i];
#pragma unroll
      for (int d = 0; d < 64; ++d) Vs[d * LDV + t] = tmp.s[d];
    }
    wait_async0();
    __syncthreads();

    // S = Q K^T (wave's 16 rows x 128 keys); software-pipelined B fragments so
    // ds_loads for tile nt+1 overlap the WMMAs of tile nt.
    v16bf a0 = load_frag_a(Qs, LDQ, wave * 16, 0);
    v16bf a1 = load_frag_a(Qs, LDQ, wave * 16, 32);
    v16bf kb0[2], kb1[2];
    kb0[0] = load_frag_b(Ks, LDQ, 0, 0);
    kb1[0] = load_frag_b(Ks, LDQ, 0, 32);
    v8f s[8];
#pragma unroll
    for (int nt = 0; nt < 8; ++nt) {
      if (nt + 1 < 8) {
        kb0[(nt + 1) & 1] = load_frag_b(Ks, LDQ, (nt + 1) * 16, 0);
        kb1[(nt + 1) & 1] = load_frag_b(Ks, LDQ, (nt + 1) * 16, 32);
      }
      v8f sc = {};
      sc = __builtin_amdgcn_wmma_f32_16x16x32_bf16(false, a0, false, kb0[nt & 1], (short)0, sc, false, false);
      sc = __builtin_amdgcn_wmma_f32_16x16x32_bf16(false, a1, false, kb1[nt & 1], (short)0, sc, false, false);
      s[nt] = sc;
    }

    // scale + causal mask + row max
    float rm[8];
#pragma unroll
    for (int v = 0; v < 8; ++v) rm[v] = -1e30f;
#pragma unroll
    for (int nt = 0; nt < 8; ++nt) {
      int kcol = kb * 128 + nt * 16 + nn;
#pragma unroll
      for (int v = 0; v < 8; ++v) {
        int qrow = q0 + wave * 16 + half * 8 + v;
        float val = s[nt][v] * 0.125f;
        if (kcol > qrow) val = -1e30f;
        s[nt][v] = val;
        rm[v] = fmaxf(rm[v], val);
      }
    }
#pragma unroll
    for (int v = 0; v < 8; ++v)
#pragma unroll
      for (int m = 1; m < 16; m <<= 1) rm[v] = fmaxf(rm[v], __shfl_xor(rm[v], m, 32));

    float corr[8], rs[8];
#pragma unroll
    for (int v = 0; v < 8; ++v) {
      float mnew = fmaxf(mrun[v], rm[v]);
      corr[v] = __expf(mrun[v] - mnew);
      mrun[v] = mnew;
      rs[v] = 0.0f;
    }
    // P = exp(S - m), write to LDS in A-fragment-readable layout (same-wave, LDS in-order)
#pragma unroll
    for (int nt = 0; nt < 8; ++nt)
#pragma unroll
      for (int v = 0; v < 8; ++v) {
        float p = __expf(s[nt][v] - mrun[v]);
        rs[v] += p;
        Ps[(wave * 16 + half * 8 + v) * LDV + nt * 16 + nn] = f2bf(p);
      }
#pragma unroll
    for (int v = 0; v < 8; ++v) {
#pragma unroll
      for (int m = 1; m < 16; m <<= 1) rs[v] += __shfl_xor(rs[v], m, 32);
      lsum[v] = lsum[v] * corr[v] + rs[v];
    }
#pragma unroll
    for (int j = 0; j < 4; ++j)
#pragma unroll
      for (int v = 0; v < 8; ++v) o[j][v] *= corr[v];

    // O += P * V  (K-dim 128 -> 4 steps), pipelined A (from Ps) and B (from Vs) fragments
    v16bf ap[2], bv[2];
    ap[0] = load_frag_a(Ps, LDV, wave * 16, 0);
    bv[0] = load_frag_b(Vs, LDV, 0, 0);
#pragma unroll
    for (int ks = 0; ks < 4; ++ks) {
      if (ks + 1 < 4) ap[(ks + 1) & 1] = load_frag_a(Ps, LDV, wave * 16, (ks + 1) * 32);
#pragma unroll
      for (int j = 0; j < 4; ++j) {
        int nj = ks * 4 + j + 1;           // next (ks,j) fragment index
        if (nj < 16) bv[(j + 1) & 1] = load_frag_b(Vs, LDV, (nj & 3) * 16, (nj >> 2) * 32);
        o[j] = __builtin_amdgcn_wmma_f32_16x16x32_bf16(
            false, ap[ks & 1], false, bv[j & 1], (short)0, o[j], false, false);
      }
    }
  }

  // normalize + store bf16 y
#pragma unroll
  for (int v = 0; v < 8; ++v) lsum[v] = 1.0f / lsum[v];
#pragma unroll
  for (int j = 0; j < 4; ++j)
#pragma unroll
    for (int v = 0; v < 8; ++v) {
      int qrow = q0 + wave * 16 + half * 8 + v;
      Y[base + (size_t)qrow * 1024 + j * 16 + nn] = f2bf(o[j][v] * lsum[v]);
    }
}

// ---------------- host launch ----------------

extern "C" void kernel_launch(void* const* d_in, const int* in_sizes, int n_in,
                              void* d_out, int out_size, void* d_ws, size_t ws_size,
                              hipStream_t stream) {
  const float* x  = (const float*)d_in[0];
  const float* Wk = (const float*)d_in[1];
  const float* bk = (const float*)d_in[2];
  const float* Wq = (const float*)d_in[3];
  const float* bq = (const float*)d_in[4];
  const float* Wv = (const float*)d_in[5];
  const float* bv = (const float*)d_in[6];
  const float* Wp = (const float*)d_in[7];
  const float* bp = (const float*)d_in[8];
  float* out = (float*)d_out;

  const size_t SZX = (size_t)4096 * 1024;   // B*T*C
  const size_t SZW = (size_t)1024 * 1024;
  ush* ws   = (ush*)d_ws;
  ush* x_bf = ws;
  ush* Wqt  = ws + SZX;
  ush* Wkt  = Wqt + SZW;
  ush* Wvt  = Wkt + SZW;
  ush* Wpt  = Wvt + SZW;
  ush* q_bf = Wpt + SZW;
  ush* k_bf = q_bf + SZX;
  ush* v_bf = k_bf + SZX;
  ush* y_bf = v_bf + SZX;

  float* y_out = out;            // outputs: y, k_cache, v_cache (flat, fp32)
  float* k_out = out + SZX;
  float* v_out = out + 2 * SZX;

  cvt_f32_bf16<<<dim3(8192), dim3(256), 0, stream>>>(x, x_bf, (int)SZX);
  transpose_cvt<<<dim3(4096), dim3(256), 0, stream>>>(Wq, Wqt);
  transpose_cvt<<<dim3(4096), dim3(256), 0, stream>>>(Wk, Wkt);
  transpose_cvt<<<dim3(4096), dim3(256), 0, stream>>>(Wv, Wvt);
  transpose_cvt<<<dim3(4096), dim3(256), 0, stream>>>(Wp, Wpt);

  dim3 ggrid(1024 / 128, 4096 / 128);
  gemm_bf16<false, true><<<ggrid, 256, 0, stream>>>(x_bf, Wqt, bq, nullptr, q_bf);
  gemm_bf16<true,  true><<<ggrid, 256, 0, stream>>>(x_bf, Wkt, bk, k_out,  k_bf);
  gemm_bf16<true,  true><<<ggrid, 256, 0, stream>>>(x_bf, Wvt, bv, v_out,  v_bf);

  attn_kernel<<<dim3(32, 16, 2), dim3(128), 0, stream>>>(q_bf, k_bf, v_bf, y_bf);

  gemm_bf16<true, false><<<ggrid, 256, 0, stream>>>(y_bf, Wpt, bp, y_out, nullptr);
}
